// GroupedQuaryAttention_66202625900874
// MI455X (gfx1250) — matrix operand order repaired and checked by reading, not compiled, not run on hardware
//
#include <hip/hip_runtime.h>
#include <hip/hip_bf16.h>

#define DI __device__ __forceinline__

typedef __bf16 bf16_t;
typedef __attribute__((ext_vector_type(16))) __bf16 v16bf;
typedef __attribute__((ext_vector_type(8)))  __bf16 v8bf;
typedef __attribute__((ext_vector_type(8)))  float   v8f;

union AFrag { v16bf v; v8bf h[2]; };

static constexpr int Bc   = 2;
static constexpr int Sc   = 2048;
static constexpr int Dc   = 2048;
static constexpr int HQc  = 32;
static constexpr int HKVc = 8;
static constexpr int HDc  = 64;
// G = HQ/HKV = 4

DI bf16_t f2bf(float f) {
  union { float f; unsigned u; } i; i.f = f;
  unsigned u = i.u;
  unsigned short s = (unsigned short)((u + 0x7FFFu + ((u >> 16) & 1u)) >> 16);
  union { unsigned short s; bf16_t b; } o; o.s = s;
  return o.b;
}

// Async global -> LDS copy, 16 bytes per lane (ASYNCcnt-tracked, no VGPR data).
DI void async_stage_b128(unsigned lds_byte_addr, const void* gaddr) {
  asm volatile("global_load_async_to_lds_b128 %0, %1, off"
               :: "v"(lds_byte_addr), "v"(gaddr) : "memory");
}
DI void async_wait0() {
  asm volatile("s_wait_asynccnt 0x0" ::: "memory");
}

DI v8f wmma_bf16(const AFrag& a, const AFrag& b, v8f c) {
  return __builtin_amdgcn_wmma_f32_16x16x32_bf16(
      false, a.v, false, b.v, (short)0, c, false, false);
}

// ---------------------------------------------------------------- cast kernels
__global__ void cast_f32_bf16_kernel(const float* __restrict__ in,
                                     bf16_t* __restrict__ out, int n) {
  int i = blockIdx.x * blockDim.x + threadIdx.x;
  if (i < n) out[i] = f2bf(in[i]);
}

// out[n*K + k] = bf16(in[k*N + n])   (weight transpose: [K][N] -> [N][K])
__global__ void transpose_cast_kernel(const float* __restrict__ in,
                                      bf16_t* __restrict__ out, int K, int N) {
  int i = blockIdx.x * blockDim.x + threadIdx.x;
  if (i < K * N) {
    int n = i / K, k = i - n * K;
    out[i] = f2bf(in[(size_t)k * N + n]);
  }
}

// ---------------------------------------------------------------- bf16 GEMM
// One 32-deep K step: 2 A-fragments (rows m..m+31) x 4 B-fragments (64 cols)
// = 8 WMMAs, B-fragments read from the given LDS buffer.
DI void gemm_step(const bf16_t* bsbuf, int ln, int half,
                  const bf16_t* arow0, const bf16_t* arow1, int k0,
                  v8f (&c)[2][4]) {
  AFrag a0, a1;
  a0.h[0] = *(const v8bf*)(arow0 + k0 + half * 8);
  a0.h[1] = *(const v8bf*)(arow0 + k0 + 16 + half * 8);
  a1.h[0] = *(const v8bf*)(arow1 + k0 + half * 8);
  a1.h[1] = *(const v8bf*)(arow1 + k0 + 16 + half * 8);
  __builtin_prefetch(arow0 + k0 + 128, 0, 1);
  __builtin_prefetch(arow1 + k0 + 128, 0, 1);
#pragma unroll
  for (int j = 0; j < 4; ++j) {
    AFrag bfr;
    const bf16_t* bp = bsbuf + (j * 16 + ln) * 32 + half * 8;
    bfr.h[0] = *(const v8bf*)(bp);
    bfr.h[1] = *(const v8bf*)(bp + 16);
    c[0][j] = wmma_bf16(a0, bfr, c[0][j]);
    c[1][j] = wmma_bf16(a1, bfr, c[1][j]);
  }
}

// C[m][n] = (sum_k A[m][k] * Bt[n][k] + bias[n]) * scale
// Block: 256 threads = 8 waves; wave tile = 32x64; block tile = 256x64.
// B tile (64n x 32k, 4 KB) double-buffered in LDS via async global->LDS loads
// overlapped with WMMA; steady-state loop is branch-free (manual 2x unroll,
// compile-time buffer selection), with a peeled tail. Requires K % 64 == 0.
// out_mode: 0 = bf16 [M][N], 1 = bf16 scattered as Vt[b][kvh][hd][s], 2 = f32 [M][N]
__global__ __launch_bounds__(256) void gemm_bf16_wmma(
    const bf16_t* __restrict__ A,    // [M][K]
    const bf16_t* __restrict__ Bt,   // [N][K]
    const float*  __restrict__ bias, // [N]
    void* __restrict__ out,
    int M, int N, int K, float scale, int out_mode) {
  __shared__ bf16_t Bs[2][64 * 32];

  const int tid  = threadIdx.x;
  const int wave = tid >> 5;
  const int lane = tid & 31;
  const int ln   = lane & 15;
  const int half = lane >> 4;
  const int m0   = blockIdx.x * 256 + wave * 32;
  const int n0   = blockIdx.y * 64;

  const bf16_t* arow0 = A + (size_t)(m0 + ln) * K;
  const bf16_t* arow1 = arow0 + (size_t)16 * K;

  // staging assignment: thread -> (n_local, k_part) covering 64x32 tile in 16 B units
  const int n_l = tid >> 2;
  const int kp  = (tid & 3) * 8;
  const bf16_t* bsrc = Bt + (size_t)(n0 + n_l) * K + kp;
  const unsigned lds0 = (unsigned)(size_t)&Bs[0][n_l * 32 + kp];
  const unsigned lds1 = (unsigned)(size_t)&Bs[1][n_l * 32 + kp];

  v8f c[2][4] = {};

  // prologue: stage first B tile into buf0
  async_stage_b128(lds0, bsrc);
  async_wait0();
  __syncthreads();

  // steady state: 64 deep per iteration, no branches, no runtime buffer select
  int k0 = 0;
  for (; k0 < K - 64; k0 += 64) {
    async_stage_b128(lds1, bsrc + (k0 + 32));
    gemm_step(Bs[0], ln, half, arow0, arow1, k0, c);
    async_wait0();
    __syncthreads();

    async_stage_b128(lds0, bsrc + (k0 + 64));
    gemm_step(Bs[1], ln, half, arow0, arow1, k0 + 32, c);
    async_wait0();
    __syncthreads();
  }
  // tail: k0 == K - 64; buf0 holds tile k0
  async_stage_b128(lds1, bsrc + (k0 + 32));
  gemm_step(Bs[0], ln, half, arow0, arow1, k0, c);
  async_wait0();
  __syncthreads();
  gemm_step(Bs[1], ln, half, arow0, arow1, k0 + 32, c);

  // Epilogue: D layout lane ln = n, VGPR r -> row r + 8*half (per 16-row subtile)
#pragma unroll
  for (int i = 0; i < 2; ++i) {
#pragma unroll
    for (int j = 0; j < 4; ++j) {
      int n = n0 + j * 16 + ln;
      float bn = bias ? bias[n] : 0.0f;
#pragma unroll
      for (int r = 0; r < 8; ++r) {
        int m = m0 + i * 16 + r + half * 8;
        float val = (c[i][j][r] + bn) * scale;
        if (out_mode == 2) {
          ((float*)out)[(size_t)m * N + n] = val;
        } else if (out_mode == 0) {
          ((bf16_t*)out)[(size_t)m * N + n] = f2bf(val);
        } else {
          // Vt[b][kvh][hd][s] ; m = b*S + s ; n = kvh*64 + hd
          int bb = m >> 11;            // m / S
          int s  = m & (Sc - 1);
          ((bf16_t*)out)[(((size_t)bb * HKVc + (n >> 6)) * HDc + (n & 63)) * Sc + s] =
              f2bf(val);
        }
      }
    }
  }
}

// ---------------------------------------------------------------- flash attention
// 4 waves per block; each wave owns one (b, head, 16-row q tile).
// Qp pre-scaled by 1/sqrt(HD).
__global__ __launch_bounds__(128) void flash_gqa_wmma(
    const bf16_t* __restrict__ Qp,   // [B,S,HQ,HD]
    const bf16_t* __restrict__ Kp,   // [B,S,HKV,HD]
    const bf16_t* __restrict__ Vt,   // [B,HKV,HD,S]
    const float*  __restrict__ mask, // [B,1,S,S]
    bf16_t* __restrict__ attn) {     // [B,S,HQ*HD]
  __shared__ bf16_t Ps[4][16 * 32];  // per-wave P tile, row-major [m][k]

  const int wave = threadIdx.x >> 5;
  const int lane = threadIdx.x & 31;
  const int ln   = lane & 15;
  const int half = lane >> 4;
  const int q0   = (blockIdx.x * 4 + wave) * 16;
  const int h    = blockIdx.y;
  const int b    = blockIdx.z;
  const int kvh  = h >> 2;           // h / G
  bf16_t* Pw = Ps[wave];

  // Q A-fragments for the two 32-wide halves of HD=64
  AFrag aq0, aq1;
  {
    const bf16_t* qrow = Qp + (((size_t)(b * Sc + q0 + ln)) * HQc + h) * HDc;
    aq0.h[0] = *(const v8bf*)(qrow + half * 16);
    aq0.h[1] = *(const v8bf*)(qrow + half * 16 + 8);
    aq1.h[0] = *(const v8bf*)(qrow + 32 + half * 16);
    aq1.h[1] = *(const v8bf*)(qrow + 32 + half * 16 + 8);
  }

  float mrow[8], lrow[8], alpha[8];
  v8f acc[4] = {};
#pragma unroll
  for (int r = 0; r < 8; ++r) { mrow[r] = -1e30f; lrow[r] = 0.0f; }

  const float* mbase = mask + (size_t)b * Sc * Sc;

  for (int kt = 0; kt < Sc; kt += 32) {
    v8f sc[2];
#pragma unroll
    for (int t = 0; t < 2; ++t) {
      int kb = kt + t * 16;
      const bf16_t* krow = Kp + (((size_t)(b * Sc + kb + ln)) * HKVc + kvh) * HDc;
      AFrag bk0, bk1;   // B = K^T: lane ln = key column, 16 contiguous hd
      bk0.h[0] = *(const v8bf*)(krow + half * 16);
      bk0.h[1] = *(const v8bf*)(krow + half * 16 + 8);
      bk1.h[0] = *(const v8bf*)(krow + 32 + half * 16);
      bk1.h[1] = *(const v8bf*)(krow + 32 + half * 16 + 8);
      v8f cs = {};
      cs = wmma_bf16(aq0, bk0, cs);
      cs = wmma_bf16(aq1, bk1, cs);
      // scores already include 1/sqrt(HD); apply mask bias: - (1/mask + 1)
#pragma unroll
      for (int r = 0; r < 8; ++r) {
        float mv = mbase[(size_t)(q0 + r + half * 8) * Sc + kb + ln];
        cs[r] -= (1.0f / mv + 1.0f);
      }
      sc[t] = cs;
    }

    // Online softmax per row (rows live within 16-lane halves)
#pragma unroll
    for (int r = 0; r < 8; ++r) {
      float vmax = fmaxf(sc[0][r], sc[1][r]);
      vmax = fmaxf(vmax, __shfl_xor(vmax, 1, 32));
      vmax = fmaxf(vmax, __shfl_xor(vmax, 2, 32));
      vmax = fmaxf(vmax, __shfl_xor(vmax, 4, 32));
      vmax = fmaxf(vmax, __shfl_xor(vmax, 8, 32));
      float nm = fmaxf(mrow[r], vmax);
      float a  = __expf(mrow[r] - nm);
      float p0 = __expf(sc[0][r] - nm);
      float p1 = __expf(sc[1][r] - nm);
      sc[0][r] = p0;
      sc[1][r] = p1;
      float ps = p0 + p1;
      ps += __shfl_xor(ps, 1, 32);
      ps += __shfl_xor(ps, 2, 32);
      ps += __shfl_xor(ps, 4, 32);
      ps += __shfl_xor(ps, 8, 32);
      lrow[r]  = a * lrow[r] + ps;
      mrow[r]  = nm;
      alpha[r] = a;
    }
#pragma unroll
    for (int ch = 0; ch < 4; ++ch)
#pragma unroll
      for (int r = 0; r < 8; ++r) acc[ch][r] *= alpha[r];

    // Re-layout P: C-fragment -> LDS row-major -> A-fragment
#pragma unroll
    for (int r = 0; r < 8; ++r) {
      int m = r + half * 8;
      Pw[m * 32 + ln]      = f2bf(sc[0][r]);
      Pw[m * 32 + 16 + ln] = f2bf(sc[1][r]);
    }
    __syncthreads();
    AFrag ap;
    ap.h[0] = *(const v8bf*)(&Pw[ln * 32 + half * 8]);
    ap.h[1] = *(const v8bf*)(&Pw[ln * 32 + 16 + half * 8]);

    // O += P(16x32) x V(32x16) per 16-wide hd chunk; Vt gives contiguous keys
#pragma unroll
    for (int ch = 0; ch < 4; ++ch) {
      const bf16_t* vp =
          Vt + (((size_t)b * HKVc + kvh) * HDc + ch * 16 + ln) * Sc + kt + half * 8;
      AFrag bv;
      bv.h[0] = *(const v8bf*)(vp);
      bv.h[1] = *(const v8bf*)(vp + 16);
      acc[ch] = wmma_bf16(ap, bv, acc[ch]);
    }
  }

  // Finalize: divide by softmax denominator, store bf16 attn output
#pragma unroll
  for (int ch = 0; ch < 4; ++ch) {
#pragma unroll
    for (int r = 0; r < 8; ++r) {
      int m = q0 + r + half * 8;
      float o = acc[ch][r] / lrow[r];
      attn[((size_t)(b * Sc + m)) * (HQc * HDc) + h * HDc + ch * 16 + ln] = f2bf(o);
    }
  }
}

// ---------------------------------------------------------------- host launch
extern "C" void kernel_launch(void* const* d_in, const int* in_sizes, int n_in,
                              void* d_out, int out_size, void* d_ws, size_t ws_size,
                              hipStream_t stream) {
  const float* q    = (const float*)d_in[0];
  const float* k    = (const float*)d_in[1];
  const float* v    = (const float*)d_in[2];
  const float* mask = (const float*)d_in[3];
  const float* Wq   = (const float*)d_in[4];
  const float* bq   = (const float*)d_in[5];
  const float* Wk   = (const float*)d_in[6];
  const float* bk   = (const float*)d_in[7];
  const float* Wv   = (const float*)d_in[8];
  const float* bv   = (const float*)d_in[9];
  const float* Wo   = (const float*)d_in[10];
  const float* bo   = (const float*)d_in[11];

  const int M   = Bc * Sc;          // 4096
  const int K   = Dc;               // 2048
  const int Nq  = HQc * HDc;        // 2048
  const int Nkv = HKVc * HDc;       // 512

  char* ws = (char*)d_ws;
  size_t off = 0;
  auto alloc = [&](size_t bytes) {
    size_t o = off;
    off = (off + bytes + 255) & ~(size_t)255;
    return o;
  };
  bf16_t* qb   = (bf16_t*)(ws + alloc((size_t)M * K * 2));
  bf16_t* kb   = (bf16_t*)(ws + alloc((size_t)M * K * 2));
  bf16_t* vb   = (bf16_t*)(ws + alloc((size_t)M * K * 2));
  bf16_t* WqT  = (bf16_t*)(ws + alloc((size_t)K * Nq * 2));
  bf16_t* WkT  = (bf16_t*)(ws + alloc((size_t)K * Nkv * 2));
  bf16_t* WvT  = (bf16_t*)(ws + alloc((size_t)K * Nkv * 2));
  bf16_t* WoT  = (bf16_t*)(ws + alloc((size_t)Nq * Dc * 2));
  bf16_t* Qp   = (bf16_t*)(ws + alloc((size_t)M * Nq * 2));
  bf16_t* Kp   = (bf16_t*)(ws + alloc((size_t)M * Nkv * 2));
  bf16_t* Vt   = (bf16_t*)(ws + alloc((size_t)M * Nkv * 2));
  bf16_t* attn = (bf16_t*)(ws + alloc((size_t)M * Nq * 2));

  const int nElem = M * K;  // 8,388,608
  cast_f32_bf16_kernel<<<(nElem + 255) / 256, 256, 0, stream>>>(q, qb, nElem);
  cast_f32_bf16_kernel<<<(nElem + 255) / 256, 256, 0, stream>>>(k, kb, nElem);
  cast_f32_bf16_kernel<<<(nElem + 255) / 256, 256, 0, stream>>>(v, vb, nElem);

  transpose_cast_kernel<<<((K * Nq) + 255) / 256, 256, 0, stream>>>(Wq, WqT, K, Nq);
  transpose_cast_kernel<<<((K * Nkv) + 255) / 256, 256, 0, stream>>>(Wk, WkT, K, Nkv);
  transpose_cast_kernel<<<((K * Nkv) + 255) / 256, 256, 0, stream>>>(Wv, WvT, K, Nkv);
  transpose_cast_kernel<<<((Nq * Dc) + 255) / 256, 256, 0, stream>>>(Wo, WoT, Nq, Dc);

  const float qscale = 0.125f;  // 1/sqrt(HD=64)
  gemm_bf16_wmma<<<dim3(M / 256, Nq / 64), 256, 0, stream>>>(
      qb, WqT, bq, (void*)Qp, M, Nq, K, qscale, 0);
  gemm_bf16_wmma<<<dim3(M / 256, Nkv / 64), 256, 0, stream>>>(
      kb, WkT, bk, (void*)Kp, M, Nkv, K, 1.0f, 0);
  gemm_bf16_wmma<<<dim3(M / 256, Nkv / 64), 256, 0, stream>>>(
      vb, WvT, bv, (void*)Vt, M, Nkv, K, 1.0f, 1);

  flash_gqa_wmma<<<dim3(Sc / 64, HQc, Bc), 128, 0, stream>>>(Qp, Kp, Vt, mask, attn);

  gemm_bf16_wmma<<<dim3(M / 256, Dc / 64), 256, 0, stream>>>(
      attn, WoT, bo, d_out, M, Dc, Nq, 1.0f, 2);
}
